// GPT_27857157882041
// MI455X (gfx1250) — matrix operand order repaired and checked by reading, not compiled, not run on hardware
//
#include <hip/hip_runtime.h>
#include <cstdint>

// ---------------- model constants ----------------
constexpr int Tt = 2048;
constexpr int Dd = 768;
constexpr int Hh = 12;
constexpr int HD = 64;
constexpr int FF = 3072;
constexpr int Vv = 50257;
constexpr int VvPad = 50304;   // Vv rounded up to multiple of 128
constexpr int Ll = 2;

typedef _Float16 f16;
typedef __attribute__((ext_vector_type(16))) _Float16 v16h;
typedef __attribute__((ext_vector_type(8)))  float    v8f;

union Frag16 { v16h v; unsigned int u[8]; unsigned short s[16]; };

// ------- fp32 [K][N] -> f16 transposed [Nalloc][K], zero-padded rows -------
__global__ __launch_bounds__(256) void cvtT(const float* __restrict__ src,
                                            f16* __restrict__ dst,
                                            int K, int N, int Nalloc) {
  __shared__ float tile[32][33];
  const int tid = threadIdx.x;
  const int tx = tid & 31;
  const int ty = tid >> 5;
  const int nb = blockIdx.x * 32;
  const int kb = blockIdx.y * 32;
#pragma unroll
  for (int r = ty; r < 32; r += 8) {
    int k = kb + r, n = nb + tx;
    tile[r][tx] = (k < K && n < N) ? src[(size_t)k * N + n] : 0.f;
  }
  __syncthreads();
#pragma unroll
  for (int r = ty; r < 32; r += 8) {
    int n = nb + r, k = kb + tx;
    if (n < Nalloc && k < K) dst[(size_t)n * K + k] = (f16)tile[tx][r];
  }
}

// ---------------- embedding ----------------
__global__ __launch_bounds__(256) void embed_kernel(const int* __restrict__ idx,
                                                    const float* __restrict__ tok,
                                                    const float* __restrict__ pos,
                                                    float* __restrict__ x) {
  int t = blockIdx.x;
  int tokid = idx[t];
  const float* tr = tok + (size_t)tokid * Dd;
  const float* pr = pos + (size_t)t * Dd;
  float* xr = x + (size_t)t * Dd;
  for (int c = threadIdx.x; c < Dd; c += 256) xr[c] = tr[c] + pr[c];
}

// ---------------- layernorm: fp32 in, f16 out ----------------
__global__ __launch_bounds__(256) void ln_kernel(const float* __restrict__ x,
                                                 const float* __restrict__ w,
                                                 const float* __restrict__ b,
                                                 f16* __restrict__ out) {
  __shared__ float red[256];
  const int tid = threadIdx.x;
  const int row = blockIdx.x;
  const float* xr = x + (size_t)row * Dd;

  float s = 0.f;
  for (int c = tid; c < Dd; c += 256) s += xr[c];
  red[tid] = s; __syncthreads();
  for (int st = 128; st > 0; st >>= 1) {
    if (tid < st) red[tid] += red[tid + st];
    __syncthreads();
  }
  float mu = red[0] / (float)Dd;
  __syncthreads();

  float v = 0.f;
  for (int c = tid; c < Dd; c += 256) { float d = xr[c] - mu; v += d * d; }
  red[tid] = v; __syncthreads();
  for (int st = 128; st > 0; st >>= 1) {
    if (tid < st) red[tid] += red[tid + st];
    __syncthreads();
  }
  float rstd = rsqrtf(red[0] / (float)Dd + 1e-5f);

  f16* orow = out + (size_t)row * Dd;
  for (int c = tid; c < Dd; c += 256)
    orow[c] = (f16)((xr[c] - mu) * rstd * w[c] + b[c]);
}

// ---------------- WMMA GEMM, async-LDS double-buffered & pipelined ----------
// C[M,N] = A[M,K](f16 row-major) @ BT[Npad,K](f16 row-major, pre-transposed,
// zero-padded). block = 256 threads (8 waves), tile 64(M) x 128(N), K step 32.
// Each wave: 32x32 output = 2 A-frags x 2 B-frags -> 4 WMMAs per K-step.
// Every thread stages 2x b64 (A) + 2x b128 (B) async chunks per slab
// -> exactly 4 ASYNCcnt per wave -> partial wait `s_wait_asynccnt 4` keeps the
// lookahead slab in flight while WMMAs run.
#define BM 64
#define BN 128
#define BK 32

__global__ __launch_bounds__(256) void gemm16(
    const f16* __restrict__ A, const f16* __restrict__ BT,
    const float* __restrict__ bias, const float* __restrict__ res,
    float* __restrict__ outF, f16* __restrict__ out16,
    int M, int N, int K, int gelu) {
  __shared__ f16 As[2][BM * BK];   // 2 x 4 KB
  __shared__ f16 Bs[2][BN * BK];   // 2 x 8 KB

  const int tid = threadIdx.x;
  const int lane = tid & 31;
  const int wave = tid >> 5;
  const int mt = wave >> 2;        // 0..1 : 32-row slab
  const int nt = wave & 3;         // 0..3 : 32-col slab
  const int m0 = blockIdx.y * BM;
  const int n0 = blockIdx.x * BN;
  const int hi = lane >> 4;
  const int ln = lane & 15;

  v8f acc[2][2];
#pragma unroll
  for (int mi = 0; mi < 2; ++mi)
#pragma unroll
    for (int ni = 0; ni < 2; ++ni)
      acc[mi][ni] = (v8f){0.f,0.f,0.f,0.f,0.f,0.f,0.f,0.f};

  // stage one 32-wide K slab into LDS buffer `buf` (uniform 4 asyncs / wave)
  auto stage = [&](int k0, int buf) {
    // A tile: 64 rows x 64 B = 512 chunks of 8 B (2 per thread)
#pragma unroll
    for (int c2 = 0; c2 < 2; ++c2) {
      int c = tid + c2 * 256;
      int row = c >> 3, cc = c & 7;
      unsigned long long ga = (unsigned long long)(uintptr_t)(
          A + (size_t)(m0 + row) * K + k0 + cc * 4);
      unsigned la = (unsigned)(uintptr_t)(&As[buf][c * 4]);
      asm volatile("global_load_async_to_lds_b64 %0, %1, off"
                   :: "v"(la), "v"(ga) : "memory");
    }
    // B tile: 128 rows x 64 B = 512 chunks of 16 B (2 per thread, no guard:
    // BT is padded/zero-filled to a multiple of 128 rows)
#pragma unroll
    for (int c2 = 0; c2 < 2; ++c2) {
      int c = tid + c2 * 256;
      int nrow = c >> 2, cc = c & 3;
      unsigned long long gb = (unsigned long long)(uintptr_t)(
          BT + (size_t)(n0 + nrow) * K + k0 + cc * 8);
      unsigned lb = (unsigned)(uintptr_t)(&Bs[buf][c * 8]);
      asm volatile("global_load_async_to_lds_b128 %0, %1, off"
                   :: "v"(lb), "v"(gb) : "memory");
    }
  };

  stage(0, 0);

  const int nk = K >> 5;
  for (int kt = 0; kt < nk; ++kt) {
    const int buf = kt & 1;
    if (kt + 1 < nk) {
      stage((kt + 1) << 5, buf ^ 1);                    // lookahead slab
      asm volatile("s_wait_asynccnt 0x4" ::: "memory"); // slab kt landed,
    } else {                                            // slab kt+1 in flight
      asm volatile("s_wait_asynccnt 0x0" ::: "memory");
    }
    __syncthreads();    // all waves' copies for slab kt visible

    const unsigned int* Asb = reinterpret_cast<const unsigned int*>(&As[buf][0]);
    const unsigned int* Bsb = reinterpret_cast<const unsigned int*>(&Bs[buf][0]);

    Frag16 af[2], bf[2];
#pragma unroll
    for (int mi = 0; mi < 2; ++mi) {
      const int am = mt * 32 + mi * 16 + ln;
#pragma unroll
      for (int p = 0; p < 8; ++p) { // A layout: kk = (p<4?0:16)+hi*8+2*(p&3)
        int kk = ((p & 4) ? 16 : 0) + hi * 8 + (p & 3) * 2;
        af[mi].u[p] = Asb[am * 16 + (kk >> 1)];
      }
    }
#pragma unroll
    for (int ni = 0; ni < 2; ++ni) {
      const int bn = nt * 32 + ni * 16 + ln;
#pragma unroll
      for (int p = 0; p < 8; ++p)   // B layout: kk = (p<4?0:16)+hi*8+2*(p&3)
        bf[ni].u[p] = Bsb[bn * 16 + ((p & 4) ? 8 : 0) + hi * 4 + (p & 3)];
    }
#pragma unroll
    for (int mi = 0; mi < 2; ++mi)
#pragma unroll
      for (int ni = 0; ni < 2; ++ni)
        acc[mi][ni] = __builtin_amdgcn_wmma_f32_16x16x32_f16(
            false, af[mi].v, false, bf[ni].v, (short)0, acc[mi][ni],
            false, false);

    __syncthreads();    // reads of buf done before slab kt+2 overwrites it
  }

  // epilogue: C/D layout -> lane ln = col, row = r + 8*hi
#pragma unroll
  for (int mi = 0; mi < 2; ++mi)
#pragma unroll
    for (int ni = 0; ni < 2; ++ni) {
      int gn = n0 + nt * 32 + ni * 16 + ln;
      if (gn < N) {
        float bv = bias ? bias[gn] : 0.f;
#pragma unroll
        for (int r = 0; r < 8; ++r) {
          int gm = m0 + mt * 32 + mi * 16 + r + 8 * hi;
          float v = acc[mi][ni][r] + bv;
          if (res)  v += res[(size_t)gm * N + gn];
          if (gelu) v = 0.5f * v * (1.0f + erff(v * 0.70710678118f));
          if (outF)  outF[(size_t)gm * N + gn] = v;
          if (out16) out16[(size_t)gm * N + gn] = (f16)v;
        }
      }
    }
}

// ---------------- V transpose per head: qkv16 -> VT[h][d][t] ----------------
__global__ __launch_bounds__(256) void vtrans_kernel(const f16* __restrict__ qkv,
                                                     f16* __restrict__ vt) {
  int t = blockIdx.x;
  const f16* vr = qkv + (size_t)t * 3 * Dd + 2 * Dd;
  for (int c = threadIdx.x; c < Dd; c += 256)
    vt[(size_t)c * Tt + t] = vr[c];      // c = h*64 + d
}

// ---------------- flash-style causal attention ----------------
// grid (Tt/16, Hh), 32 threads (1 wave). qkv f16 [Tt][3*Dd] (bias included).
__global__ __launch_bounds__(32) void attn_kernel(const f16* __restrict__ qkv,
                                                  const f16* __restrict__ vt,
                                                  f16* __restrict__ out) {
  __shared__ unsigned int Ps[16 * 16];   // 16x32 f16 P tile

  const int lane = threadIdx.x & 31;
  const int hi = lane >> 4;
  const int ln = lane & 15;
  const int qb = blockIdx.x;
  const int h = blockIdx.y;
  const int LDQ = 3 * Dd;
  const int qoff = h * HD;
  const int koff = Dd + h * HD;

  Frag16 qf[2];
  {
    const int m = qb * 16 + ln;
    const unsigned int* qrow =
        reinterpret_cast<const unsigned int*>(qkv + (size_t)m * LDQ + qoff);
#pragma unroll
    for (int half = 0; half < 2; ++half)
#pragma unroll
      for (int p = 0; p < 8; ++p) {
        int kk = half * 32 + ((p & 4) ? 16 : 0) + hi * 8 + (p & 3) * 2;
        qf[half].u[p] = qrow[kk >> 1];
      }
  }

  v8f o[4];
#pragma unroll
  for (int t = 0; t < 4; ++t) o[t] = (v8f){0.f,0.f,0.f,0.f,0.f,0.f,0.f,0.f};
  float rm[8], rs[8];
#pragma unroll
  for (int r = 0; r < 8; ++r) { rm[r] = -1e30f; rs[r] = 0.f; }

  const int qmax = qb * 16 + 15;
  const int nsteps = (qmax >> 5) + 1;

  for (int j = 0; j < nsteps; ++j) {
    const int s0 = j * 32;
    // ---- S = Q @ K^T ----
    v8f sc[2];
#pragma unroll
    for (int ct = 0; ct < 2; ++ct) {
      v8f s = {0.f,0.f,0.f,0.f,0.f,0.f,0.f,0.f};
      const int key = s0 + ct * 16 + ln;
      const unsigned int* krow =
          reinterpret_cast<const unsigned int*>(qkv + (size_t)key * LDQ + koff);
#pragma unroll
      for (int half = 0; half < 2; ++half) {
        Frag16 bf;
#pragma unroll
        for (int p = 0; p < 8; ++p) {
          int kk = half * 32 + ((p & 4) ? 16 : 0) + hi * 8 + (p & 3) * 2;
          bf.u[p] = krow[kk >> 1];
        }
        s = __builtin_amdgcn_wmma_f32_16x16x32_f16(false, qf[half].v, false,
                                                   bf.v, (short)0, s, false, false);
      }
      sc[ct] = s;
    }
    // ---- scale + causal mask ----
#pragma unroll
    for (int ct = 0; ct < 2; ++ct) {
      int col = s0 + ct * 16 + ln;
#pragma unroll
      for (int r = 0; r < 8; ++r) {
        int row = qb * 16 + r + 8 * hi;
        float v = sc[ct][r] * 0.125f;
        sc[ct][r] = (col <= row) ? v : -1e30f;
      }
    }
    // ---- online softmax ----
#pragma unroll
    for (int r = 0; r < 8; ++r) {
      float v = fmaxf(sc[0][r], sc[1][r]);
#pragma unroll
      for (int off = 8; off >= 1; off >>= 1) v = fmaxf(v, __shfl_xor(v, off, 32));
      float mn = fmaxf(rm[r], v);
      float so = __expf(rm[r] - mn);
      sc[0][r] = __expf(sc[0][r] - mn);
      sc[1][r] = __expf(sc[1][r] - mn);
      float ps = sc[0][r] + sc[1][r];
#pragma unroll
      for (int off = 8; off >= 1; off >>= 1) ps += __shfl_xor(ps, off, 32);
      rs[r] = rs[r] * so + ps;
      rm[r] = mn;
      o[0][r] *= so; o[1][r] *= so; o[2][r] *= so; o[3][r] *= so;
    }
    // ---- P (C layout) -> LDS -> A fragment ----
    f16* Pld = reinterpret_cast<f16*>(Ps);
#pragma unroll
    for (int ct = 0; ct < 2; ++ct)
#pragma unroll
      for (int r = 0; r < 8; ++r)
        Pld[(r + 8 * hi) * 32 + ct * 16 + ln] = (f16)sc[ct][r];
    __syncthreads();
    Frag16 pf;
#pragma unroll
    for (int p = 0; p < 8; ++p) {
      int kk = ((p & 4) ? 16 : 0) + hi * 8 + (p & 3) * 2;
      pf.u[p] = Ps[ln * 16 + (kk >> 1)];
    }
    // ---- O += P @ V : VT[h][d][t] makes key-pairs contiguous dwords ----
#pragma unroll
    for (int t = 0; t < 4; ++t) {
      const unsigned int* vrow = reinterpret_cast<const unsigned int*>(
          vt + (size_t)(h * HD + t * 16 + ln) * Tt);
      Frag16 vf;
#pragma unroll
      for (int p = 0; p < 8; ++p)
        vf.u[p] = vrow[(s0 >> 1) + ((p & 4) ? 8 : 0) + hi * 4 + (p & 3)];
      o[t] = __builtin_amdgcn_wmma_f32_16x16x32_f16(false, pf.v, false, vf.v,
                                                    (short)0, o[t], false, false);
    }
    __syncthreads();
  }

#pragma unroll
  for (int t = 0; t < 4; ++t)
#pragma unroll
    for (int r = 0; r < 8; ++r) {
      int m = qb * 16 + r + 8 * hi;
      float v = o[t][r] / rs[r];
      out[(size_t)m * Dd + h * HD + t * 16 + ln] = (f16)v;
    }
}

// ---------------- host-side orchestration ----------------
extern "C" void kernel_launch(void* const* d_in, const int* in_sizes, int n_in,
                              void* d_out, int out_size, void* d_ws, size_t ws_size,
                              hipStream_t stream) {
  (void)in_sizes; (void)n_in; (void)out_size; (void)ws_size;

  const int*   idx     = (const int*)d_in[0];
  const float* tok_emb = (const float*)d_in[1];
  const float* pos_emb = (const float*)d_in[2];
  const float* ln1_w   = (const float*)d_in[3];
  const float* ln1_b   = (const float*)d_in[4];
  const float* qkv_w   = (const float*)d_in[5];
  const float* qkv_b   = (const float*)d_in[6];
  const float* proj_w  = (const float*)d_in[7];
  const float* proj_b  = (const float*)d_in[8];
  const float* ln2_w   = (const float*)d_in[9];
  const float* ln2_b   = (const float*)d_in[10];
  const float* fc1_w   = (const float*)d_in[11];
  const float* fc1_b   = (const float*)d_in[12];
  const float* fc2_w   = (const float*)d_in[13];
  const float* fc2_b   = (const float*)d_in[14];
  const float* lnf_w   = (const float*)d_in[15];
  const float* lnf_b   = (const float*)d_in[16];
  const float* head_w  = (const float*)d_in[17];
  float* logits = (float*)d_out;

  char* ws = (char*)d_ws;
  size_t off = 0;
  auto take = [&](size_t bytes) -> void* {
    void* p = ws + off;
    off += (bytes + 255) & ~(size_t)255;
    return p;
  };
  float* xf32  = (float*)take((size_t)Tt * Dd * 4);
  f16* qkv16   = (f16*)take((size_t)Tt * 3 * Dd * 2);
  f16* actA    = (f16*)take((size_t)Tt * FF * 2);
  f16* actB    = (f16*)take((size_t)Tt * FF * 2);
  f16* vt16    = (f16*)take((size_t)Tt * Dd * 2);
  f16* w_qkv   = (f16*)take((size_t)Ll * Dd * 3 * Dd * 2);   // [3D][D] per layer
  f16* w_proj  = (f16*)take((size_t)Ll * Dd * Dd * 2);       // [D][D]
  f16* w_fc1   = (f16*)take((size_t)Ll * Dd * FF * 2);       // [FF][D]
  f16* w_fc2   = (f16*)take((size_t)Ll * FF * Dd * 2);       // [D][FF]
  f16* w_head  = (f16*)take((size_t)Dd * VvPad * 2);         // [VvPad][D]

  // weight convert + transpose (deterministic, every call)
  auto t32 = [](int n) { return (unsigned)((n + 31) / 32); };
  for (int l = 0; l < Ll; ++l) {
    cvtT<<<dim3(t32(3 * Dd), t32(Dd)), 256, 0, stream>>>(
        qkv_w + (size_t)l * Dd * 3 * Dd, w_qkv + (size_t)l * Dd * 3 * Dd,
        Dd, 3 * Dd, 3 * Dd);
    cvtT<<<dim3(t32(Dd), t32(Dd)), 256, 0, stream>>>(
        proj_w + (size_t)l * Dd * Dd, w_proj + (size_t)l * Dd * Dd, Dd, Dd, Dd);
    cvtT<<<dim3(t32(FF), t32(Dd)), 256, 0, stream>>>(
        fc1_w + (size_t)l * Dd * FF, w_fc1 + (size_t)l * Dd * FF, Dd, FF, FF);
    cvtT<<<dim3(t32(Dd), t32(FF)), 256, 0, stream>>>(
        fc2_w + (size_t)l * FF * Dd, w_fc2 + (size_t)l * FF * Dd, FF, Dd, Dd);
  }
  cvtT<<<dim3(t32(VvPad), t32(Dd)), 256, 0, stream>>>(head_w, w_head,
                                                      Dd, Vv, VvPad);

  embed_kernel<<<Tt, 256, 0, stream>>>(idx, tok_emb, pos_emb, xf32);

  const dim3 gQKV((3 * Dd + BN - 1) / BN, Tt / BM);
  const dim3 gPRJ((Dd + BN - 1) / BN, Tt / BM);
  const dim3 gFC1((FF + BN - 1) / BN, Tt / BM);
  const dim3 gHED((Vv + BN - 1) / BN, Tt / BM);

  for (int l = 0; l < Ll; ++l) {
    ln_kernel<<<Tt, 256, 0, stream>>>(xf32, ln1_w + l * Dd, ln1_b + l * Dd, actA);
    gemm16<<<gQKV, 256, 0, stream>>>(actA, w_qkv + (size_t)l * Dd * 3 * Dd,
                                     qkv_b + (size_t)l * 3 * Dd, nullptr,
                                     nullptr, qkv16, Tt, 3 * Dd, Dd, 0);
    vtrans_kernel<<<Tt, 256, 0, stream>>>(qkv16, vt16);
    attn_kernel<<<dim3(Tt / 16, Hh), 32, 0, stream>>>(qkv16, vt16, actA);
    gemm16<<<gPRJ, 256, 0, stream>>>(actA, w_proj + (size_t)l * Dd * Dd,
                                     proj_b + (size_t)l * Dd, xf32,
                                     xf32, nullptr, Tt, Dd, Dd, 0);
    ln_kernel<<<Tt, 256, 0, stream>>>(xf32, ln2_w + l * Dd, ln2_b + l * Dd, actA);
    gemm16<<<gFC1, 256, 0, stream>>>(actA, w_fc1 + (size_t)l * Dd * FF,
                                     fc1_b + (size_t)l * FF, nullptr,
                                     nullptr, actB, Tt, FF, Dd, 1);
    gemm16<<<gPRJ, 256, 0, stream>>>(actB, w_fc2 + (size_t)l * FF * Dd,
                                     fc2_b + (size_t)l * Dd, xf32,
                                     xf32, nullptr, Tt, Dd, FF, 0);
  }

  ln_kernel<<<Tt, 256, 0, stream>>>(xf32, lnf_w, lnf_b, actA);
  gemm16<<<gHED, 256, 0, stream>>>(actA, w_head, nullptr, nullptr,
                                   logits, nullptr, Tt, Vv, Dd, 0);
}
